// MultiQueryAttention_20375324852774
// MI455X (gfx1250) — matrix operand order repaired and checked by reading, not compiled, not run on hardware
//
#include <hip/hip_runtime.h>
#include <cstdint>

// MQA forward for MI455X (gfx1250): bf16 WMMA everywhere, flash-style attention,
// double-buffered async global->LDS staging (ASYNCcnt path) in the GEMMs.
// B=2, S=2048, D=2048, H=16, HD=128.

#define BDIM 2
#define SDIM 2048
#define DDIM 2048
#define HHEADS 16
#define HD 128

typedef __attribute__((ext_vector_type(16))) __bf16 v16bf;
typedef __attribute__((ext_vector_type(8))) float v8f;

union FragU { v16bf v; unsigned u[8]; };
union Vec8 { uint4 u; unsigned short s[8]; };

__device__ inline v8f vzero8() {
  v8f z = {0.f, 0.f, 0.f, 0.f, 0.f, 0.f, 0.f, 0.f};
  return z;
}

__device__ inline unsigned short f2bf(float x) {  // round-to-nearest-even
  unsigned u = __float_as_uint(x);
  u += 0x7FFFu + ((u >> 16) & 1u);
  return (unsigned short)(u >> 16);
}

__device__ inline v8f wmma_bf16(v16bf a, v16bf b, v8f c) {
  // v_wmma_f32_16x16x32_bf16  D = A*B + C
  return __builtin_amdgcn_wmma_f32_16x16x32_bf16(false, a, false, b, (short)0, c,
                                                 false, false);
}

// Async copy: 16 bytes per lane, global (SGPR base + 32-bit lane offset) -> LDS.
// Tracked by ASYNCcnt; wave-level completion via s_wait_asynccnt.
__device__ inline void async_b128(const void* gbase, unsigned goff_bytes,
                                  unsigned lds_addr) {
  asm volatile("global_load_async_to_lds_b128 %0, %1, %2"
               :: "v"(lds_addr), "v"(goff_bytes), "s"(gbase)
               : "memory");
}
__device__ inline void wait_async0() {
  asm volatile("s_wait_asynccnt 0x0" ::: "memory");
}

// A fragment: 16x32 (MxK) bf16, source row-major with given element stride.
// ISA layout: lane half h, m = lane&15; VGPR v: k = (v>>2)*16 + h*8 + (v&3)*2.
__device__ inline v16bf frag_a(const unsigned short* s, int row0, int k0, int stride) {
  const int lane = threadIdx.x & 31;
  const int h = lane >> 4;
  const int m = row0 + (lane & 15);
  FragU f;
#pragma unroll
  for (int v = 0; v < 8; ++v) {
    int k = k0 + (v >> 2) * 16 + h * 8 + (v & 3) * 2;
    f.u[v] = *(const unsigned*)(s + (size_t)m * stride + k);
  }
  return f.v;
}

// B fragment: 32x16 (KxN) bf16, source stored as NxK row-major (i.e. B^T).
// ISA layout: lane half h, n = lane&15; VGPR v: k = (v>>2)*16 + h*8 + (v&3)*2.
__device__ inline v16bf frag_b(const unsigned short* s, int n0, int k0, int stride) {
  const int lane = threadIdx.x & 31;
  const int h = lane >> 4;
  const int n = n0 + (lane & 15);
  FragU f;
#pragma unroll
  for (int v = 0; v < 8; ++v) {
    int k = k0 + (v >> 2) * 16 + h * 8 + (v & 3) * 2;
    f.u[v] = *(const unsigned*)(s + (size_t)n * stride + k);
  }
  return f.v;
}

// ---------------------------------------------------------------- conversions
__global__ __launch_bounds__(256) void cvt_f32_to_bf16(const float* __restrict__ in,
                                                       unsigned short* __restrict__ out,
                                                       int n) {
  int i = blockIdx.x * 256 + threadIdx.x;
  if (i < n) out[i] = f2bf(in[i]);
}

// ------------------------------------------------------------------- GEMM
// Out[M,N] = A[M,K](bf16) @ Bw[N,K]^T(bf16) + bias[N]; Out bf16 or f32.
// Block 128x128, K-step 32, 8 waves each computing 32x64 (2x4 WMMA tiles).
// Double-buffered: async global->LDS copies for tile kt+1 overlap WMMAs on kt.
template <bool OUT_BF16>
__global__ __launch_bounds__(256) void gemm_bf16_xwt(const unsigned short* __restrict__ A,
                                                     const unsigned short* __restrict__ Bw,
                                                     const float* __restrict__ bias,
                                                     void* __restrict__ Out,
                                                     int M, int N, int K) {
  __shared__ __align__(16) unsigned short sA[2][128 * 40];  // 128x32, stride 40
  __shared__ __align__(16) unsigned short sB[2][128 * 40];
  const int wid = threadIdx.x >> 5;
  const int wm = wid & 3;   // 4 wave rows  -> 32 rows each
  const int wn = wid >> 2;  // 2 wave cols  -> 64 cols each
  const int m_blk = blockIdx.y * 128;
  const int n_blk = blockIdx.x * 128;

  const unsigned sA_base = (unsigned)(uintptr_t)&sA[0][0];
  const unsigned sB_base = (unsigned)(uintptr_t)&sB[0][0];

  auto stage = [&](int buf, int k0) {
#pragma unroll
    for (int i = 0; i < 2; ++i) {  // 256 thr * 2 * 16B = one 128x32 bf16 tile
      int idx = threadIdx.x + i * 256;
      int row = idx >> 2;
      int cg = (idx & 3) * 8;
      unsigned loff = (unsigned)((buf * 128 * 40 + row * 40 + cg) * 2);
      async_b128(A, (unsigned)(((size_t)(m_blk + row) * K + k0 + cg) * 2),
                 sA_base + loff);
      async_b128(Bw, (unsigned)(((size_t)(n_blk + row) * K + k0 + cg) * 2),
                 sB_base + loff);
    }
  };

  v8f acc[2][4];
#pragma unroll
  for (int i = 0; i < 2; ++i)
#pragma unroll
    for (int j = 0; j < 4; ++j) acc[i][j] = vzero8();

  const int nt = K >> 5;
  stage(0, 0);  // prologue
  for (int kt = 0; kt < nt; ++kt) {
    const int cur = kt & 1;
    wait_async0();        // our copies for buf[cur] have landed
    __syncthreads();      // everyone's copies landed; buf[cur^1] free to refill
    if (kt + 1 < nt) stage(cur ^ 1, (kt + 1) * 32);

    const unsigned short* curA = &sA[cur][0];
    const unsigned short* curB = &sB[cur][0];
    v16bf af[2];
#pragma unroll
    for (int i = 0; i < 2; ++i) af[i] = frag_a(curA, wm * 32 + i * 16, 0, 40);
#pragma unroll
    for (int j = 0; j < 4; ++j) {
      v16bf bf = frag_b(curB, wn * 64 + j * 16, 0, 40);
#pragma unroll
      for (int i = 0; i < 2; ++i) acc[i][j] = wmma_bf16(af[i], bf, acc[i][j]);
    }
  }

  const int lane = threadIdx.x & 31;
  const int hlf = lane >> 4;
  const int ln = lane & 15;
#pragma unroll
  for (int j = 0; j < 4; ++j) {
    int n = n_blk + wn * 64 + j * 16 + ln;
    float bv = bias[n];
#pragma unroll
    for (int i = 0; i < 2; ++i) {
#pragma unroll
      for (int r = 0; r < 8; ++r) {  // C layout: row = r + 8*half
        int m = m_blk + wm * 32 + i * 16 + hlf * 8 + r;
        float val = acc[i][j][r] + bv;
        if (OUT_BF16)
          ((unsigned short*)Out)[(size_t)m * N + n] = f2bf(val);
        else
          ((float*)Out)[(size_t)m * N + n] = val;
      }
    }
  }
}

// --------------------------------------------------------------- attention
// One WG per (b, h, 128-query block). 8 waves; each wave owns 16 full query
// rows (row reductions stay inside a lane-half). Key blocks of 64, online
// softmax, O accumulated in registers via WMMA C-operand. K block staged with
// async global->LDS copies; V transposed through VGPRs (needs the shuffle).
__global__ __launch_bounds__(256) void mqa_flash(const unsigned short* __restrict__ q,
                                                 const unsigned short* __restrict__ kbuf,
                                                 const unsigned short* __restrict__ vbuf,
                                                 unsigned short* __restrict__ obuf) {
  __shared__ __align__(16) unsigned short sK[64 * 136];   // t x d (NxK form)
  __shared__ __align__(16) unsigned short sVt[128 * 68];  // V transposed: d x t
  __shared__ __align__(16) unsigned short sP[8 * 16 * 68];  // per-wave P 16x64

  const int qblk = blockIdx.x & 15;
  const int h = (blockIdx.x >> 4) & 15;
  const int b = blockIdx.x >> 8;
  const int wid = threadIdx.x >> 5;
  const int lane = threadIdx.x & 31;
  const int hlf = lane >> 4;
  const int ln = lane & 15;

  const unsigned sK_base = (unsigned)(uintptr_t)&sK[0];

  // Q fragments for this wave's 16 rows stay in registers for all key blocks.
  const unsigned short* qbase =
      q + ((size_t)(b * SDIM + qblk * 128 + wid * 16)) * DDIM + h * HD;
  v16bf qf[4];
#pragma unroll
  for (int kk = 0; kk < 4; ++kk) qf[kk] = frag_a(qbase, 0, kk * 32, DDIM);

  v8f acc_o[8];
#pragma unroll
  for (int n = 0; n < 8; ++n) acc_o[n] = vzero8();
  float mrow[8], lrow[8];
#pragma unroll
  for (int r = 0; r < 8; ++r) { mrow[r] = -1e30f; lrow[r] = 0.f; }

  const float scale = 0.08838834764831845f;  // 1/sqrt(128)

  for (int j = 0; j < SDIM / 64; ++j) {
    __syncthreads();  // previous block's consumers done with sK/sVt
    // Stage K via async copies; V via regular loads + transposed LDS stores
    // (stride 68 elems -> 136B rows => conflict-free scatter across 32 lanes).
#pragma unroll
    for (int i = 0; i < 4; ++i) {
      int idx = threadIdx.x + i * 256;
      int row = idx >> 4;          // key index 0..63
      int cg = (idx & 15) * 8;     // d base
      size_t g = (size_t)(b * SDIM + j * 64 + row) * HD + cg;
      async_b128(kbuf, (unsigned)(g * 2),
                 sK_base + (unsigned)((row * 136 + cg) * 2));
      Vec8 val;
      val.u = *(const uint4*)(vbuf + g);
#pragma unroll
      for (int e = 0; e < 8; ++e) sVt[(cg + e) * 68 + row] = val.s[e];
    }
    wait_async0();
    __syncthreads();

    // S = Q K^T : 16x64 per wave (4 n-tiles), K-dim = HD = 128 (4 steps)
    v8f sacc[4];
#pragma unroll
    for (int n = 0; n < 4; ++n) sacc[n] = vzero8();
#pragma unroll
    for (int kk = 0; kk < 4; ++kk) {
#pragma unroll
      for (int n = 0; n < 4; ++n) {
        v16bf bf = frag_b(sK, n * 16, kk * 32, 136);
        sacc[n] = wmma_bf16(qf[kk], bf, sacc[n]);
      }
    }

    // Online softmax. Lane element r = row (r + 8*half); columns = 4 tiles x
    // 16 lanes of a half => shfl_xor 1/2/4/8 completes the 64-wide row.
#pragma unroll
    for (int r = 0; r < 8; ++r) {
      float mx = -1e30f;
#pragma unroll
      for (int n = 0; n < 4; ++n) {
        sacc[n][r] *= scale;
        mx = fmaxf(mx, sacc[n][r]);
      }
      mx = fmaxf(mx, __shfl_xor(mx, 1));
      mx = fmaxf(mx, __shfl_xor(mx, 2));
      mx = fmaxf(mx, __shfl_xor(mx, 4));
      mx = fmaxf(mx, __shfl_xor(mx, 8));
      float nm = fmaxf(mrow[r], mx);
      float al = __expf(mrow[r] - nm);
      mrow[r] = nm;
      float sum = 0.f;
#pragma unroll
      for (int n = 0; n < 4; ++n) {
        float p = __expf(sacc[n][r] - nm);
        sacc[n][r] = p;
        sum += p;
      }
      sum += __shfl_xor(sum, 1);
      sum += __shfl_xor(sum, 2);
      sum += __shfl_xor(sum, 4);
      sum += __shfl_xor(sum, 8);
      lrow[r] = lrow[r] * al + sum;
#pragma unroll
      for (int n = 0; n < 8; ++n) acc_o[n][r] *= al;
    }

    // Stage P as bf16 (per-wave region; wave-local RAW handled by dscnt wait).
    unsigned short* myP = sP + wid * 16 * 68;
#pragma unroll
    for (int n = 0; n < 4; ++n)
#pragma unroll
      for (int r = 0; r < 8; ++r)
        myP[(hlf * 8 + r) * 68 + n * 16 + ln] = f2bf(sacc[n][r]);

    // O += P @ V : K = 64 (2 steps), 8 n-tiles over HD=128.
#pragma unroll
    for (int kk = 0; kk < 2; ++kk) {
      v16bf pf = frag_a(myP, 0, kk * 32, 68);
#pragma unroll
      for (int n = 0; n < 8; ++n) {
        v16bf vf = frag_b(sVt, n * 16, kk * 32, 68);
        acc_o[n] = wmma_bf16(pf, vf, acc_o[n]);
      }
    }
  }

  // Normalize and store in the reference's (b, h, s, hd) flat order.
  size_t base = (size_t)b * SDIM * DDIM + (size_t)h * SDIM * HD;
#pragma unroll
  for (int r = 0; r < 8; ++r) {
    int srow = qblk * 128 + wid * 16 + hlf * 8 + r;
    float inv = 1.0f / lrow[r];
#pragma unroll
    for (int n = 0; n < 8; ++n) {
      int d = n * 16 + ln;
      obuf[base + (size_t)srow * HD + d] = f2bf(acc_o[n][r] * inv);
    }
  }
}

// ------------------------------------------------------------------ launch
extern "C" void kernel_launch(void* const* d_in, const int* in_sizes, int n_in,
                              void* d_out, int out_size, void* d_ws, size_t ws_size,
                              hipStream_t stream) {
  (void)in_sizes; (void)n_in; (void)out_size; (void)ws_size;
  const float* x    = (const float*)d_in[0];
  const float* Wq_w = (const float*)d_in[1];
  const float* Wq_b = (const float*)d_in[2];
  const float* Wk_w = (const float*)d_in[3];
  const float* Wk_b = (const float*)d_in[4];
  const float* Wv_w = (const float*)d_in[5];
  const float* Wv_b = (const float*)d_in[6];
  const float* Wo_w = (const float*)d_in[7];
  const float* Wo_b = (const float*)d_in[8];

  const int BS = BDIM * SDIM;  // 4096

  char* ws = (char*)d_ws;
  size_t off = 0;
  auto carve = [&](size_t bytes) {
    void* p = ws + off;
    off += (bytes + 255) & ~(size_t)255;
    return p;
  };
  unsigned short* xb  = (unsigned short*)carve((size_t)BS * DDIM * 2);
  unsigned short* wqb = (unsigned short*)carve((size_t)DDIM * DDIM * 2);
  unsigned short* wkb = (unsigned short*)carve((size_t)HD * DDIM * 2);
  unsigned short* wvb = (unsigned short*)carve((size_t)HD * DDIM * 2);
  unsigned short* wob = (unsigned short*)carve((size_t)DDIM * DDIM * 2);
  unsigned short* qb  = (unsigned short*)carve((size_t)BS * DDIM * 2);
  unsigned short* kb  = (unsigned short*)carve((size_t)BS * HD * 2);
  unsigned short* vb  = (unsigned short*)carve((size_t)BS * HD * 2);
  unsigned short* ab  = (unsigned short*)carve((size_t)BS * DDIM * 2);

  auto cvt = [&](const float* src, unsigned short* dst, int n) {
    cvt_f32_to_bf16<<<(n + 255) / 256, 256, 0, stream>>>(src, dst, n);
  };
  cvt(x, xb, BS * DDIM);
  cvt(Wq_w, wqb, DDIM * DDIM);
  cvt(Wk_w, wkb, HD * DDIM);
  cvt(Wv_w, wvb, HD * DDIM);
  cvt(Wo_w, wob, DDIM * DDIM);

  // Projections (bf16 out)
  gemm_bf16_xwt<true><<<dim3(DDIM / 128, BS / 128), 256, 0, stream>>>(
      xb, wqb, Wq_b, qb, BS, DDIM, DDIM);
  gemm_bf16_xwt<true><<<dim3(HD / 128, BS / 128), 256, 0, stream>>>(
      xb, wkb, Wk_b, kb, BS, HD, DDIM);
  gemm_bf16_xwt<true><<<dim3(HD / 128, BS / 128), 256, 0, stream>>>(
      xb, wvb, Wv_b, vb, BS, HD, DDIM);

  // Flash attention: B*H*(S/128) = 512 workgroups
  mqa_flash<<<BDIM * HHEADS * (SDIM / 128), 256, 0, stream>>>(qb, kb, vb, ab);

  // Output projection (f32 out -> d_out)
  gemm_bf16_xwt<false><<<dim3(DDIM / 128, BS / 128), 256, 0, stream>>>(
      ab, wob, Wo_b, (float*)d_out, BS, DDIM, DDIM);
}
